// TorchSchNet_16819091931689
// MI455X (gfx1250) — matrix operand order repaired and checked by reading, not compiled, not run on hardware
//
#include <hip/hip_runtime.h>
#include <hip/hip_bf16.h>

#define NNODES  50000
#define NEDGES  640000
#define HID     128
#define NF      128
#define NGAUSS  50
#define NLAYER  3
#define CUTOFF_F 5.0f
#define NGRAPHS 64
#define PI_F 3.14159265358979f

typedef __attribute__((ext_vector_type(16))) _Float16 v16h;
typedef __attribute__((ext_vector_type(8)))  float    v8f;

union V16H { v16h v; _Float16 h[16]; unsigned u[8]; };
union V8F  { v8f v; float f[8]; };

// A-fragment K index for element e (0..15) of a 16x32 f16 A tile (contiguous: k = 16*(lane/16) + e)
__device__ __forceinline__ int amapK(int e, int lane) {
  int hi = (lane & 16) ? 16 : 0;
  return hi + e;
}

__device__ __forceinline__ float sspf(float x) {
  float sp = (x > 15.0f) ? x : log1pf(expf(x));
  return sp - 0.69314718055994531f;  // shifted softplus
}

// ---------------------------------------------------------------------------
// Pack a row-major fp32 weight [Kdim x 128] into WMMA B-fragment order, f16.
// Linear layout == fragment address: dst[(((t*Ksteps)+s)*32+lane)*16 + h],
// value = W[s*32 + 16*(lane>=16) + h][t*16 + lane%16]  (zero-padded K).
// ---------------------------------------------------------------------------
__global__ void pack_b_kernel(const float* __restrict__ src,
                              _Float16* __restrict__ dst,
                              int Kdim, int Ksteps) {
  int total = Ksteps * 8 * 32 * 16;
  for (int idx = blockIdx.x * blockDim.x + threadIdx.x; idx < total;
       idx += gridDim.x * blockDim.x) {
    int hh   = idx & 15;
    int lane = (idx >> 4) & 31;
    int ts   = idx >> 9;
    int s2   = ts % Ksteps;
    int tt   = ts / Ksteps;
    int k    = s2 * 32 + ((lane & 16) ? 16 : 0) + hh;
    int col  = tt * 16 + (lane & 15);
    float v  = (k < Kdim) ? src[k * 128 + col] : 0.0f;
    dst[idx] = (_Float16)v;
  }
}

__global__ void zero_kernel(float* __restrict__ p, long n) {
  long i = (long)blockIdx.x * blockDim.x + threadIdx.x;
  if (i < n) p[i] = 0.0f;
}

__global__ void embed_kernel(const int* __restrict__ z,
                             const float* __restrict__ emb,
                             float* __restrict__ h) {
  long idx = (long)blockIdx.x * blockDim.x + threadIdx.x;
  if (idx < (long)NNODES * 128) {
    long n = idx >> 7;
    int  c = (int)(idx & 127);
    h[idx] = emb[(long)z[n] * 128 + c];
  }
}

// ---------------------------------------------------------------------------
// xf = A[N,128] @ Bpacked[128,128]   (no bias) — one 16-row tile per wave.
// ---------------------------------------------------------------------------
__global__ __launch_bounds__(256) void node_gemm_kernel(
    const float* __restrict__ A, const _Float16* __restrict__ Bp,
    float* __restrict__ out, int ntiles) {
  const int lane = threadIdx.x & 31;
  const int wid  = threadIdx.x >> 5;
  const long tile = (long)blockIdx.x * 8 + wid;
  if (tile >= ntiles) return;
  const int mrow = lane & 15;
  const int mhi  = (lane & 16) ? 8 : 0;
  V8F acc[8];
#pragma unroll
  for (int t = 0; t < 8; ++t)
#pragma unroll
    for (int r = 0; r < 8; ++r) acc[t].f[r] = 0.0f;

  const float* rowbase = A + (tile * 16 + mrow) * 128;
#pragma unroll
  for (int ks = 0; ks < 4; ++ks) {
    V16H a;
#pragma unroll
    for (int p = 0; p < 8; ++p) {
      int K = ks * 32 + amapK(2 * p, lane);
      a.h[2 * p]     = (_Float16)rowbase[K];
      a.h[2 * p + 1] = (_Float16)rowbase[K + 1];
    }
#pragma unroll
    for (int t = 0; t < 8; ++t) {
      V16H b;
      b.v = *(const v16h*)(Bp + (size_t)((t * 4 + ks) * 32 + lane) * 16);
      acc[t].v = __builtin_amdgcn_wmma_f32_16x16x32_f16(
          false, a.v, false, b.v, (short)0, acc[t].v, false, false);
    }
  }
#pragma unroll
  for (int t = 0; t < 8; ++t)
#pragma unroll
    for (int r = 0; r < 8; ++r)
      out[(tile * 16 + r + mhi) * 128 + t * 16 + mrow] = acc[t].f[r];
}

// ---------------------------------------------------------------------------
// Fused edge pipeline per 16-edge tile (one wave):
//   gaussians(+envelope) -> GEMM1(w1) -> +b1,ssp -> LDS relayout ->
//   GEMM2(w2) -> +b2, *cosine -> msg = xf[src]*W -> atomicAdd agg[dst]
// ---------------------------------------------------------------------------
__global__ __launch_bounds__(256) void edge_kernel(
    const float* __restrict__ pos, const long long* __restrict__ ei,
    const float* __restrict__ xf,
    const _Float16* __restrict__ w1p, const _Float16* __restrict__ w2p,
    const float* __restrict__ b1, const float* __restrict__ b2,
    float* __restrict__ agg) {
  __shared__ __align__(16) _Float16 ldsA[8 * 16 * 128];
  __shared__ float ldsC[8 * 16];
  __shared__ int   ldsS[8 * 16];
  __shared__ int   ldsD[8 * 16];

  const int lane = threadIdx.x & 31;
  const int wid  = threadIdx.x >> 5;
  const int mrow = lane & 15;
  const int mhi  = (lane & 16) ? 8 : 0;
  const long tile = (long)blockIdx.x * 8 + wid;
  const long e = tile * 16 + mrow;

  const int sn = (int)ei[e];
  const int dn = (int)ei[(long)NEDGES + e];
  float dx = pos[3 * dn + 0] - pos[3 * sn + 0];
  float dy = pos[3 * dn + 1] - pos[3 * sn + 1];
  float dz = pos[3 * dn + 2] - pos[3 * sn + 2];
  float d = sqrtf(dx * dx + dy * dy + dz * dz);

  const float delta = CUTOFF_F / (float)(NGAUSS - 1);
  const float coeff = -0.5f / (delta * delta);
  float srel = d * (1.0f / CUTOFF_F);
  float env = 0.0f;
  if (srel < 1.0f) {
    float dnm = 1.0f - srel * srel;
    env = expf(1.0f - 1.0f / dnm);
  }
  float Ccut = 0.5f * (cosf(d * (PI_F / CUTOFF_F)) + 1.0f);

  if (lane < 16) {
    ldsC[wid * 16 + mrow] = Ccut;
    ldsS[wid * 16 + mrow] = sn;
    ldsD[wid * 16 + mrow] = dn;
  }

  // ---- GEMM1: edge_attr(16x64, synthesized in-register) @ w1(64x128) ----
  V8F acc1[8];
#pragma unroll
  for (int t = 0; t < 8; ++t)
#pragma unroll
    for (int r = 0; r < 8; ++r) acc1[t].f[r] = 0.0f;

#pragma unroll
  for (int ks = 0; ks < 2; ++ks) {
    V16H a;
#pragma unroll
    for (int el = 0; el < 16; ++el) {
      int K = ks * 32 + amapK(el, lane);
      float g = 0.0f;
      if (K < NGAUSS) {
        float dd = d - (float)K * delta;
        g = expf(coeff * dd * dd) * env;
      }
      a.h[el] = (_Float16)g;
    }
#pragma unroll
    for (int t = 0; t < 8; ++t) {
      V16H b;
      b.v = *(const v16h*)(w1p + (size_t)((t * 2 + ks) * 32 + lane) * 16);
      acc1[t].v = __builtin_amdgcn_wmma_f32_16x16x32_f16(
          false, a.v, false, b.v, (short)0, acc1[t].v, false, false);
    }
  }

  // ---- +b1, ssp, stage D-layout -> row-major f16 tile in LDS ----
  _Float16* myA = ldsA + wid * 16 * 128;
#pragma unroll
  for (int t = 0; t < 8; ++t)
#pragma unroll
    for (int r = 0; r < 8; ++r) {
      int col = t * 16 + mrow;
      float x = acc1[t].f[r] + b1[col];
      myA[(r + mhi) * 128 + col] = (_Float16)sspf(x);
    }
  asm volatile("s_wait_dscnt 0" ::: "memory");

  // ---- GEMM2: T(16x128) @ w2(128x128) ----
  V8F acc2[8];
#pragma unroll
  for (int t = 0; t < 8; ++t)
#pragma unroll
    for (int r = 0; r < 8; ++r) acc2[t].f[r] = 0.0f;

#pragma unroll
  for (int ks = 0; ks < 4; ++ks) {
    V16H a;
    const unsigned* rowp = (const unsigned*)(myA + mrow * 128 + ks * 32);
#pragma unroll
    for (int p = 0; p < 8; ++p) {
      int K = amapK(2 * p, lane);      // even, < 32
      a.u[p] = rowp[K >> 1];
    }
#pragma unroll
    for (int t = 0; t < 8; ++t) {
      V16H b;
      b.v = *(const v16h*)(w2p + (size_t)((t * 4 + ks) * 32 + lane) * 16);
      acc2[t].v = __builtin_amdgcn_wmma_f32_16x16x32_f16(
          false, a.v, false, b.v, (short)0, acc2[t].v, false, false);
    }
  }

  // ---- W = (acc2+b2)*C ; msg = xf[src]*W ; scatter-add into agg[dst] ----
#pragma unroll
  for (int t = 0; t < 8; ++t)
#pragma unroll
    for (int r = 0; r < 8; ++r) {
      int col = t * 16 + mrow;
      int m   = r + mhi;
      float W = (acc2[t].f[r] + b2[col]) * ldsC[wid * 16 + m];
      int srcn = ldsS[wid * 16 + m];
      int dstn = ldsD[wid * 16 + m];
      float msg = W * xf[(long)srcn * 128 + col];
      atomicAdd(&agg[(long)dstn * 128 + col], msg);
    }
}

// ---------------------------------------------------------------------------
// h += ( ssp(agg @ conv2_w + c2b) @ lin_w + lnb )
// ---------------------------------------------------------------------------
__global__ __launch_bounds__(256) void node_update_kernel(
    const float* __restrict__ agg, const _Float16* __restrict__ c2p,
    const float* __restrict__ c2b, const _Float16* __restrict__ lnp,
    const float* __restrict__ lnb, float* __restrict__ h, int ntiles) {
  __shared__ __align__(16) _Float16 ldsA[8 * 16 * 128];
  const int lane = threadIdx.x & 31;
  const int wid  = threadIdx.x >> 5;
  const long tile = (long)blockIdx.x * 8 + wid;
  if (tile >= ntiles) return;
  const int mrow = lane & 15;
  const int mhi  = (lane & 16) ? 8 : 0;

  V8F acc1[8];
#pragma unroll
  for (int t = 0; t < 8; ++t)
#pragma unroll
    for (int r = 0; r < 8; ++r) acc1[t].f[r] = 0.0f;

  const float* rowbase = agg + (tile * 16 + mrow) * 128;
#pragma unroll
  for (int ks = 0; ks < 4; ++ks) {
    V16H a;
#pragma unroll
    for (int p = 0; p < 8; ++p) {
      int K = ks * 32 + amapK(2 * p, lane);
      a.h[2 * p]     = (_Float16)rowbase[K];
      a.h[2 * p + 1] = (_Float16)rowbase[K + 1];
    }
#pragma unroll
    for (int t = 0; t < 8; ++t) {
      V16H b;
      b.v = *(const v16h*)(c2p + (size_t)((t * 4 + ks) * 32 + lane) * 16);
      acc1[t].v = __builtin_amdgcn_wmma_f32_16x16x32_f16(
          false, a.v, false, b.v, (short)0, acc1[t].v, false, false);
    }
  }

  _Float16* myA = ldsA + wid * 16 * 128;
#pragma unroll
  for (int t = 0; t < 8; ++t)
#pragma unroll
    for (int r = 0; r < 8; ++r) {
      int col = t * 16 + mrow;
      float x = acc1[t].f[r] + c2b[col];
      myA[(r + mhi) * 128 + col] = (_Float16)sspf(x);
    }
  asm volatile("s_wait_dscnt 0" ::: "memory");

  V8F acc2[8];
#pragma unroll
  for (int t = 0; t < 8; ++t)
#pragma unroll
    for (int r = 0; r < 8; ++r) acc2[t].f[r] = 0.0f;

#pragma unroll
  for (int ks = 0; ks < 4; ++ks) {
    V16H a;
    const unsigned* rowp = (const unsigned*)(myA + mrow * 128 + ks * 32);
#pragma unroll
    for (int p = 0; p < 8; ++p) {
      int K = amapK(2 * p, lane);
      a.u[p] = rowp[K >> 1];
    }
#pragma unroll
    for (int t = 0; t < 8; ++t) {
      V16H b;
      b.v = *(const v16h*)(lnp + (size_t)((t * 4 + ks) * 32 + lane) * 16);
      acc2[t].v = __builtin_amdgcn_wmma_f32_16x16x32_f16(
          false, a.v, false, b.v, (short)0, acc2[t].v, false, false);
    }
  }

#pragma unroll
  for (int t = 0; t < 8; ++t)
#pragma unroll
    for (int r = 0; r < 8; ++r) {
      int col = t * 16 + mrow;
      long idx = (tile * 16 + r + mhi) * 128 + col;
      h[idx] = h[idx] + acc2[t].f[r] + lnb[col];
    }
}

// ---------------------------------------------------------------------------
// Output head: e = ssp(h @ out1_w + b1) @ out2_w + b2 ; segment-sum by graph.
// One wave per node; lane covers columns {lane, lane+32} of the 64.
// ---------------------------------------------------------------------------
__global__ __launch_bounds__(256) void head_kernel(
    const float* __restrict__ h, const float* __restrict__ w1,
    const float* __restrict__ b1, const float* __restrict__ w2,
    const float* __restrict__ b2, const int* __restrict__ batch,
    float* __restrict__ out) {
  const int lane = threadIdx.x & 31;
  const int wid  = threadIdx.x >> 5;
  const long n = (long)blockIdx.x * 8 + wid;
  if (n >= NNODES) return;
  const int c0 = lane, c1 = lane + 32;
  float s0 = b1[c0], s1 = b1[c1];
  const float* hr = h + n * 128;
#pragma unroll 4
  for (int k = 0; k < 128; ++k) {
    float hv = hr[k];
    s0 += hv * w1[k * 64 + c0];
    s1 += hv * w1[k * 64 + c1];
  }
  float v = sspf(s0) * w2[c0] + sspf(s1) * w2[c1];
#pragma unroll
  for (int off = 16; off > 0; off >>= 1) v += __shfl_xor(v, off);
  if (lane == 0) atomicAdd(&out[batch[n]], v + b2[0]);
}

// ---------------------------------------------------------------------------
extern "C" void kernel_launch(void* const* d_in, const int* in_sizes, int n_in,
                              void* d_out, int out_size, void* d_ws,
                              size_t ws_size, hipStream_t stream) {
  const int*       z       = (const int*)d_in[0];
  const float*     pos     = (const float*)d_in[1];
  const long long* ei      = (const long long*)d_in[2];
  const int*       batch   = (const int*)d_in[3];
  const float*     emb     = (const float*)d_in[4];
  const float*     mlp_w1  = (const float*)d_in[5];
  const float*     mlp_b1  = (const float*)d_in[6];
  const float*     mlp_w2  = (const float*)d_in[7];
  const float*     mlp_b2  = (const float*)d_in[8];
  const float*     conv1_w = (const float*)d_in[9];
  const float*     conv2_w = (const float*)d_in[10];
  const float*     conv2_b = (const float*)d_in[11];
  const float*     lin_w   = (const float*)d_in[12];
  const float*     lin_b   = (const float*)d_in[13];
  const float*     out1_w  = (const float*)d_in[14];
  const float*     out1_b  = (const float*)d_in[15];
  const float*     out2_w  = (const float*)d_in[16];
  const float*     out2_b  = (const float*)d_in[17];

  char* ws = (char*)d_ws;
  const size_t NH = (size_t)NNODES * 128 * sizeof(float);  // 25.6 MB
  float* h   = (float*)(ws);
  float* xf  = (float*)(ws + NH);
  float* agg = (float*)(ws + 2 * NH);
  _Float16* w1p = (_Float16*)(ws + 3 * NH);   // 3 * 8192 halfs
  _Float16* w2p = w1p + 3 * 8192;             // 3 * 16384 halfs each below
  _Float16* c1p = w2p + 3 * 16384;
  _Float16* c2p = c1p + 3 * 16384;
  _Float16* lnp = c2p + 3 * 16384;

  for (int l = 0; l < NLAYER; ++l) {
    pack_b_kernel<<<32, 256, 0, stream>>>(mlp_w1 + (size_t)l * NGAUSS * NF,
                                          w1p + (size_t)l * 8192, NGAUSS, 2);
    pack_b_kernel<<<64, 256, 0, stream>>>(mlp_w2 + (size_t)l * NF * NF,
                                          w2p + (size_t)l * 16384, NF, 4);
    pack_b_kernel<<<64, 256, 0, stream>>>(conv1_w + (size_t)l * HID * NF,
                                          c1p + (size_t)l * 16384, HID, 4);
    pack_b_kernel<<<64, 256, 0, stream>>>(conv2_w + (size_t)l * NF * HID,
                                          c2p + (size_t)l * 16384, NF, 4);
    pack_b_kernel<<<64, 256, 0, stream>>>(lin_w + (size_t)l * HID * HID,
                                          lnp + (size_t)l * 16384, HID, 4);
  }

  const long nh = (long)NNODES * 128;
  embed_kernel<<<(int)((nh + 255) / 256), 256, 0, stream>>>(z, emb, h);

  const int ntiles  = NNODES / 16;       // 3125 (exact)
  const int nblocks = (ntiles + 7) / 8;  // 391
  for (int l = 0; l < NLAYER; ++l) {
    zero_kernel<<<(int)((nh + 255) / 256), 256, 0, stream>>>(agg, nh);
    node_gemm_kernel<<<nblocks, 256, 0, stream>>>(h, c1p + (size_t)l * 16384,
                                                  xf, ntiles);
    edge_kernel<<<NEDGES / 128, 256, 0, stream>>>(
        pos, ei, xf, w1p + (size_t)l * 8192, w2p + (size_t)l * 16384,
        mlp_b1 + (size_t)l * NF, mlp_b2 + (size_t)l * NF, agg);
    node_update_kernel<<<nblocks, 256, 0, stream>>>(
        agg, c2p + (size_t)l * 16384, conv2_b + (size_t)l * HID,
        lnp + (size_t)l * 16384, lin_b + (size_t)l * HID, h, ntiles);
  }

  zero_kernel<<<1, 64, 0, stream>>>((float*)d_out, NGRAPHS);
  head_kernel<<<(NNODES + 7) / 8, 256, 0, stream>>>(h, out1_w, out1_b, out2_w,
                                                    out2_b, batch,
                                                    (float*)d_out);
}
